// fA_P4MResBlock2D_81561428951508
// MI455X (gfx1250) — compile-verified, matrix operand order checked by reading, not run on hardware
//
#include <hip/hip_runtime.h>
#include <hip/hip_bf16.h>

// ---------------------------------------------------------------------------
// P4M group-equivariant residual block on MI455X (gfx1250, wave32, WMMA+TDM).
// Heavy conv = implicit GEMM (M=768, N=3136, K=6912) via v_wmma_f32_16x16x32_bf16
// with Tensor-Data-Mover weight staging into LDS (double buffered, TENSORcnt).
// ---------------------------------------------------------------------------

typedef __bf16 v16bf __attribute__((ext_vector_type(16)));
typedef float  v8f   __attribute__((ext_vector_type(8)));
typedef unsigned int v4u __attribute__((ext_vector_type(4)));
typedef int v8i __attribute__((ext_vector_type(8)));
typedef int v4i __attribute__((ext_vector_type(4)));

#define CCH   96
#define GRP   8
#define MIDC  48
#define HWD   56
#define NPIX  (HWD*HWD)          // 3136
#define BATCH 4
#define KDIM  (CCH*GRP*9)        // 6912
#define MTOT  (GRP*CCH)          // 768
#define NELEM ((size_t)BATCH*CCH*GRP*NPIX)   // 9,633,792
#define KSTAGE 64
#define NSTAGE (KDIM/KSTAGE)     // 108
#define LDS_ROW 72               // 64 k + 8 pad (bf16 elems) -> 144B rows

#if defined(__has_builtin)
#if __has_builtin(__builtin_amdgcn_tensor_load_to_lds) && \
    __has_builtin(__builtin_amdgcn_s_wait_tensorcnt)
#define HAVE_TDM 1
#endif
#endif
#ifndef HAVE_TDM
#define HAVE_TDM 0
#endif

// p4m composition table: PERM[h][g]  (precomputed from the reference)
__device__ __constant__ int d_PERM[8][8] = {
  {0,1,2,3,4,5,6,7},
  {3,0,1,2,5,6,7,4},
  {2,3,0,1,6,7,4,5},
  {1,2,3,0,7,4,5,6},
  {4,5,6,7,0,1,2,3},
  {5,6,7,4,3,0,1,2},
  {6,7,4,5,2,3,0,1},
  {7,4,5,6,1,2,3,0}};

__device__ __forceinline__ float sigmoidf(float x) {
  return 1.0f / (1.0f + __expf(-x));
}

// (kh,kw) -> source (si,sj) in original kernel for output orientation h,
// kernel size N (3 or 7).  transformed = rot90^r ( flip_if_m ( W ) )
__device__ __forceinline__ void inv_spatial(int h, int kh, int kw, int N,
                                            int& si, int& sj) {
  int m = h >> 2, r = h & 3;
  si = kh; sj = kw;
  #pragma unroll
  for (int t = 0; t < 3; ++t) {
    if (t < r) { int ns = sj; sj = (N - 1) - si; si = ns; }
  }
  if (m) sj = (N - 1) - sj;
}

// ---------------------------------------------------------------------------
// BN statistics: per-channel mean/var over (B,G,H,W)
// ---------------------------------------------------------------------------
__global__ __launch_bounds__(256) void bn_stats_kernel(
    const float* __restrict__ x, float* __restrict__ mean, float* __restrict__ var) {
  const int c = blockIdx.x;
  __shared__ float ssum[256], ssq[256];
  float s = 0.f, s2 = 0.f;
  const int per = GRP * NPIX;
  for (int idx = threadIdx.x; idx < BATCH * per; idx += 256) {
    int b = idx / per, rem = idx - b * per;
    float v = x[((size_t)(b * CCH + c) * per) + rem];
    s += v; s2 += v * v;
  }
  ssum[threadIdx.x] = s; ssq[threadIdx.x] = s2;
  __syncthreads();
  for (int off = 128; off > 0; off >>= 1) {
    if (threadIdx.x < off) {
      ssum[threadIdx.x] += ssum[threadIdx.x + off];
      ssq[threadIdx.x]  += ssq[threadIdx.x + off];
    }
    __syncthreads();
  }
  if (threadIdx.x == 0) {
    float n = (float)(BATCH * per);
    float mu = ssum[0] / n;
    mean[c] = mu;
    var[c] = ssq[0] / n - mu * mu;
  }
}

// ---------------------------------------------------------------------------
// BN apply + ReLU -> fp32 activation buffer
// ---------------------------------------------------------------------------
__global__ __launch_bounds__(256) void bn_relu_kernel(
    const float* __restrict__ x, const float* __restrict__ mean,
    const float* __restrict__ var, const float* __restrict__ gamma,
    const float* __restrict__ beta, float* __restrict__ out) {
  size_t i = (size_t)blockIdx.x * 256 + threadIdx.x;
  if (i >= NELEM) return;
  int c = (int)((i / (GRP * NPIX)) % CCH);
  float v = (x[i] - mean[c]) * rsqrtf(var[c] + 2e-05f) * gamma[c] + beta[c];
  out[i] = fmaxf(v, 0.f);
}

// ---------------------------------------------------------------------------
// Channel attention: s[b,c,g] = mean over H,W
// ---------------------------------------------------------------------------
__global__ __launch_bounds__(256) void chan_mean_kernel(
    const float* __restrict__ a, float* __restrict__ s) {
  const int bcg = blockIdx.x;
  __shared__ float red[256];
  float acc = 0.f;
  const float* base = a + (size_t)bcg * NPIX;
  for (int i = threadIdx.x; i < NPIX; i += 256) acc += base[i];
  red[threadIdx.x] = acc;
  __syncthreads();
  for (int off = 128; off > 0; off >>= 1) {
    if (threadIdx.x < off) red[threadIdx.x] += red[threadIdx.x + off];
    __syncthreads();
  }
  if (threadIdx.x == 0) s[bcg] = red[0] * (1.0f / NPIX);
}

// t[b,m,o] = relu( sum_{c,g} W1[m,c,PERM[o][g]] * s[b,c,g] )
__global__ __launch_bounds__(256) void att_t_kernel(
    const float* __restrict__ s, const float* __restrict__ W1,
    float* __restrict__ t) {
  int tid = blockIdx.x * 256 + threadIdx.x;
  if (tid >= BATCH * MIDC * GRP) return;
  int b = tid / (MIDC * GRP);
  int m = (tid / GRP) % MIDC;
  int o = tid % GRP;
  float acc = 0.f;
  for (int c = 0; c < CCH; ++c) {
    const float* w = W1 + (size_t)(m * CCH + c) * GRP;
    const float* sv = s + (size_t)(b * CCH + c) * GRP;
    #pragma unroll
    for (int g = 0; g < GRP; ++g) acc += w[d_PERM[o][g]] * sv[g];
  }
  t[tid] = fmaxf(acc, 0.f);
}

// a_ch[b,c,o] = sigmoid( sum_{m,g} W2[c,m,PERM[o][g]] * t[b,m,g] )
__global__ __launch_bounds__(256) void att_o_kernel(
    const float* __restrict__ t, const float* __restrict__ W2,
    float* __restrict__ ach) {
  int tid = blockIdx.x * 256 + threadIdx.x;
  if (tid >= BATCH * CCH * GRP) return;
  int b = tid / (CCH * GRP);
  int c = (tid / GRP) % CCH;
  int o = tid % GRP;
  float acc = 0.f;
  for (int m = 0; m < MIDC; ++m) {
    const float* w = W2 + (size_t)(c * MIDC + m) * GRP;
    const float* tv = t + (size_t)(b * MIDC + m) * GRP;
    #pragma unroll
    for (int g = 0; g < GRP; ++g) acc += w[d_PERM[o][g]] * tv[g];
  }
  ach[tid] = sigmoidf(acc);
}

// ---------------------------------------------------------------------------
// Spatial attention pooling: p[b,{mean,max},g,y,x] over channels
// ---------------------------------------------------------------------------
__global__ __launch_bounds__(256) void spat_pool_kernel(
    const float* __restrict__ a, float* __restrict__ p) {
  int tid = blockIdx.x * 256 + threadIdx.x;
  if (tid >= BATCH * GRP * NPIX) return;
  int b = tid / (GRP * NPIX);
  int rem = tid - b * (GRP * NPIX);
  float sum = 0.f, mx = -3.4e38f;
  for (int c = 0; c < CCH; ++c) {
    float v = a[((size_t)(b * CCH + c) * GRP * NPIX) + rem];
    sum += v; mx = fmaxf(mx, v);
  }
  p[((size_t)(b * 2 + 0) * GRP * NPIX) + rem] = sum * (1.0f / CCH);
  p[((size_t)(b * 2 + 1) * GRP * NPIX) + rem] = mx;
}

// 7x7 group-equivariant conv over p -> sigmoid -> a_sp[b,o,y,x]
__global__ __launch_bounds__(256) void spat_att_kernel(
    const float* __restrict__ p, const float* __restrict__ Wsp,
    float* __restrict__ asp) {
  int tid = blockIdx.x * 256 + threadIdx.x;
  if (tid >= BATCH * GRP * NPIX) return;
  int b = tid / (GRP * NPIX);
  int rem = tid - b * (GRP * NPIX);
  int o = rem / NPIX;
  int pix = rem - o * NPIX;
  int y = pix / HWD, x = pix - y * HWD;
  float acc = 0.f;
  for (int kh = 0; kh < 7; ++kh) {
    int yy = y + kh - 3;
    if ((unsigned)yy >= (unsigned)HWD) continue;
    for (int kw = 0; kw < 7; ++kw) {
      int xx = x + kw - 3;
      if ((unsigned)xx >= (unsigned)HWD) continue;
      int si, sj; inv_spatial(o, kh, kw, 7, si, sj);
      #pragma unroll
      for (int ci = 0; ci < 2; ++ci) {
        #pragma unroll
        for (int g = 0; g < GRP; ++g) {
          float pv = p[((size_t)((b * 2 + ci) * GRP + g) * NPIX) + yy * HWD + xx];
          float wv = Wsp[(size_t)(ci * GRP + d_PERM[o][g]) * 49 + si * 7 + sj];
          acc += pv * wv;
        }
      }
    }
  }
  asp[(size_t)(b * GRP + o) * NPIX + pix] = sigmoidf(acc);
}

// ---------------------------------------------------------------------------
// Gate activations by channel+spatial attention, pack to bf16
// ---------------------------------------------------------------------------
__global__ __launch_bounds__(256) void scale_pack_kernel(
    const float* __restrict__ a, const float* __restrict__ ach,
    const float* __restrict__ asp, __bf16* __restrict__ xm) {
  size_t i = (size_t)blockIdx.x * 256 + threadIdx.x;
  if (i >= NELEM) return;
  int bcg = (int)(i / NPIX);
  int pix = (int)(i - (size_t)bcg * NPIX);
  int b = bcg / (CCH * GRP);
  int g = bcg % GRP;
  float v = a[i] * ach[bcg] * asp[(size_t)(b * GRP + g) * NPIX + pix];
  xm[i] = (__bf16)v;
}

// ---------------------------------------------------------------------------
// Weight transform: Wt[(h*96+co)][ (ci*8+g)*9 + kh*3+kw ] (bf16, row-major)
// ---------------------------------------------------------------------------
__global__ __launch_bounds__(256) void wt_transform_kernel(
    const float* __restrict__ W, __bf16* __restrict__ Wt) {
  size_t tid = (size_t)blockIdx.x * 256 + threadIdx.x;
  if (tid >= (size_t)MTOT * KDIM) return;
  int m = (int)(tid / KDIM);
  int k = (int)(tid - (size_t)m * KDIM);
  int h = m / CCH, co = m % CCH;
  int ci = k / 72;
  int rr = k - ci * 72;
  int g = rr / 9;
  int kk = rr - g * 9;
  int kh = kk / 3, kw = kk - kh * 3;
  int si, sj; inv_spatial(h, kh, kw, 3, si, sj);
  float v = W[(size_t)((co * CCH + ci) * GRP + d_PERM[h][g]) * 9 + si * 3 + sj];
  Wt[tid] = (__bf16)v;
}

// ---------------------------------------------------------------------------
// TDM: issue a 64(row) x 64(k) bf16 tile DMA  Wt -> LDS, with LDS row padding
// so each tile row lands with a 72-element (144B) stride.
// D# per CDNA5 ISA ch.8: group0 = {flags, lds_addr, global_addr, type=2},
// group1 = {data_size=2B, pad 32dw-interval/4dw-amount, tensor 6912x768,
//           tile 64x64, dim0_stride 6912}.  6-arg builtin (clang-23 form).
// ---------------------------------------------------------------------------
#if HAVE_TDM
__device__ __forceinline__ void tdm_load_tile(const __bf16* Wt, int mBase,
                                              int k0, unsigned lds_off) {
  unsigned long long ga =
      (unsigned long long)(const void*)(Wt + (size_t)mBase * KDIM + k0);
  v4u g0;
  g0.x = 1u;                                   // count=1 (valid user D#)
  g0.y = lds_off;                              // lds_addr
  g0.z = (unsigned)(ga & 0xffffffffull);       // global_addr[31:0]
  g0.w = (unsigned)((ga >> 32) & 0x01ffffffull) | (2u << 30);  // [56:32]|type=2
  v8i g1;
  g1[0] = (1 << 16)      // data_size = 2 bytes
        | (1 << 20)      // pad_enable
        | (4 << 22)      // pad_interval: every 32 DWORDs (= one 128B tile row)
        | (3 << 25);     // pad_amount: 4 DWORDs (= 16B -> 72-elem row stride)
  g1[1] = (int)(((unsigned)KDIM & 0xffffu) << 16);   // tensor_dim0 lo16
  g1[2] = (int)(((unsigned)MTOT & 0xffffu) << 16);   // dim0 hi16=0 | dim1 lo16
  g1[3] = (int)(KSTAGE << 16);                       // dim1 hi16=0 | tile_dim0
  g1[4] = 64;                                        // tile_dim1 | tile_dim2=0
  g1[5] = KDIM;                                      // tensor_dim0_stride lo32
  g1[6] = 0;
  g1[7] = 0;
  v4i gz4 = {0, 0, 0, 0};
  v8i gz8 = {0, 0, 0, 0, 0, 0, 0, 0};
  __builtin_amdgcn_tensor_load_to_lds(g0, g1, gz4, gz4, gz8, 0);
}
#endif

// ---------------------------------------------------------------------------
// Implicit-GEMM group conv via WMMA bf16.
//   per batch: D[768][3136] = Wt[768][6912] x Im2col[6912][3136]
//   block = 256 thr = 8 waves (4x2) -> 64(M) x 64(N) tile, K-stage = 64.
//   A tile staged by TDM (double buffered, s_wait_tensorcnt pipeline);
//   B tile im2col-gathered, stored transposed [n][k].
// ---------------------------------------------------------------------------
struct BFPair { uint4 lo, hi; };

__global__ __launch_bounds__(256) void conv_wmma_kernel(
    const __bf16* __restrict__ xm,    // [B][96][8][56][56] bf16
    const __bf16* __restrict__ Wt,    // [768][6912] bf16
    const float*  __restrict__ resid, // residual (layer 2) or ignored
    float* __restrict__ out,          // [B][96][8][56][56] f32
    int add_residual) {
  __shared__ __align__(16) unsigned short As[2][64 * LDS_ROW];
  __shared__ __align__(16) unsigned short Bs[64 * LDS_ROW];

  const int tid  = threadIdx.x;
  const int lane = tid & 31;
  const int wave = tid >> 5;
  const int wm = wave & 3;            // M sub-tile (0..3) -> 16 rows each
  const int wn = wave >> 2;           // N super-tile (0..1) -> 32 cols each
  const int b     = blockIdx.z;
  const int mBase = blockIdx.y * 64;
  const int nBase = blockIdx.x * 64;

  // B loader: thread -> (n 0..63, 16-wide k chunk)
  const int ln = tid & 63;
  const int lk = (tid >> 6) * 16;
  const int ng = nBase + ln;
  const int pp = ng / HWD;
  const int qq = ng - pp * HWD;
  const unsigned short* xs = (const unsigned short*)xm;

  v8f acc[2];
  acc[0] = (v8f){0.f, 0.f, 0.f, 0.f, 0.f, 0.f, 0.f, 0.f};
  acc[1] = (v8f){0.f, 0.f, 0.f, 0.f, 0.f, 0.f, 0.f, 0.f};

#if HAVE_TDM
  const unsigned as_base0 = (unsigned)(unsigned long long)&As[0][0];
  const unsigned as_base1 = (unsigned)(unsigned long long)&As[1][0];
  if (wave == 0) tdm_load_tile(Wt, mBase, 0, as_base0);   // prime stage 0
#else
  const int arow = tid >> 2;          // fallback cooperative A loader
  const int akc  = (tid & 3) * 16;
  const uint4* Ap = (const uint4*)(Wt + (size_t)(mBase + arow) * KDIM);
#endif

  for (int s = 0; s < NSTAGE; ++s) {
    const int k0 = s * KSTAGE;

    // ---- stage B tile (im2col gather), transposed [n][k] ----
    #pragma unroll 4
    for (int i = 0; i < 16; ++i) {
      int k = k0 + lk + i;
      int ci = k / 72;
      int rr = k - ci * 72;
      int g = rr / 9;
      int kk = rr - g * 9;
      int kh = kk / 3, kw = kk - kh * 3;
      int yy = pp + kh - 1, xx = qq + kw - 1;
      unsigned short v = 0;
      if ((unsigned)yy < (unsigned)HWD && (unsigned)xx < (unsigned)HWD)
        v = xs[(size_t)((b * CCH + ci) * GRP + g) * NPIX + yy * HWD + xx];
      Bs[ln * LDS_ROW + lk + i] = v;
    }

#if HAVE_TDM
    if (wave == 0) {
      if (s + 1 < NSTAGE) {
        // prefetch next stage into the other buffer, then wait until only
        // that prefetch is outstanding (TENSORcnt is in-order per wave).
        tdm_load_tile(Wt, mBase, k0 + KSTAGE, (s & 1) ? as_base0 : as_base1);
        __builtin_amdgcn_s_wait_tensorcnt(1);
      } else {
        __builtin_amdgcn_s_wait_tensorcnt(0);
      }
    }
#else
    {
      uint4 a0 = Ap[(k0 + akc) >> 3];
      uint4 a1 = Ap[(k0 + akc + 8) >> 3];
      unsigned short* dst = &As[s & 1][arow * LDS_ROW + akc];
      *(uint4*)dst = a0;
      *(uint4*)(dst + 8) = a1;
    }
#endif
    __syncthreads();

    // ---- fragments per documented CDNA5 VGPR striping ----
    const unsigned short* Acur = As[s & 1];
    const int half = lane >> 4;
    const int ar = (wm * 16 + (lane & 15)) * LDS_ROW;
    const int bc0 = (wn * 32 + (lane & 15)) * LDS_ROW;
    const int bc1 = bc0 + 16 * LDS_ROW;
    #pragma unroll
    for (int ks = 0; ks < KSTAGE; ks += 32) {
      BFPair apr, bp0, bp1;
      apr.lo = *(const uint4*)&Acur[ar + ks + half * 8];
      apr.hi = *(const uint4*)&Acur[ar + ks + 16 + half * 8];
      bp0.lo = *(const uint4*)&Bs[bc0 + ks + half * 8];
      bp0.hi = *(const uint4*)&Bs[bc0 + ks + 16 + half * 8];
      bp1.lo = *(const uint4*)&Bs[bc1 + ks + half * 8];
      bp1.hi = *(const uint4*)&Bs[bc1 + ks + 16 + half * 8];
      v16bf af  = __builtin_bit_cast(v16bf, apr);
      v16bf bf0 = __builtin_bit_cast(v16bf, bp0);
      v16bf bf1 = __builtin_bit_cast(v16bf, bp1);
      acc[0] = __builtin_amdgcn_wmma_f32_16x16x32_bf16(
          false, af, false, bf0, (short)0, acc[0], false, false);
      acc[1] = __builtin_amdgcn_wmma_f32_16x16x32_bf16(
          false, af, false, bf1, (short)0, acc[1], false, false);
    }
    __syncthreads();
  }

  // ---- epilogue: scatter D tiles to (B, No, G, H, W) layout ----
  #pragma unroll
  for (int nt = 0; nt < 2; ++nt) {
    const int colg = nBase + wn * 32 + nt * 16 + (lane & 15);
    const int m0 = mBase + wm * 16 + (lane >> 4) * 8;
    #pragma unroll
    for (int r = 0; r < 8; ++r) {
      int m = m0 + r;
      int h_or = m / CCH, co = m - h_or * CCH;
      size_t oidx = (size_t)((b * CCH + co) * GRP + h_or) * NPIX + colg;
      float v = acc[nt][r];
      if (add_residual) v += resid[oidx];
      out[oidx] = v;
    }
  }
}

// ---------------------------------------------------------------------------
// Host-side orchestration
// ---------------------------------------------------------------------------
static void run_layer(const float* input, const float* gamma, const float* beta,
                      const float* W, const float* A1, const float* A2,
                      const float* Wsp, const float* resid, float* out,
                      float* a, __bf16* xm, __bf16* Wt, float* mean, float* var,
                      float* s, float* t, float* ach, float* p, float* asp,
                      int add_residual, hipStream_t stream) {
  const int eb = (int)((NELEM + 255) / 256);
  bn_stats_kernel<<<CCH, 256, 0, stream>>>(input, mean, var);
  bn_relu_kernel<<<eb, 256, 0, stream>>>(input, mean, var, gamma, beta, a);
  chan_mean_kernel<<<BATCH * CCH * GRP, 256, 0, stream>>>(a, s);
  att_t_kernel<<<(BATCH * MIDC * GRP + 255) / 256, 256, 0, stream>>>(s, A1, t);
  att_o_kernel<<<(BATCH * CCH * GRP + 255) / 256, 256, 0, stream>>>(t, A2, ach);
  spat_pool_kernel<<<(BATCH * GRP * NPIX + 255) / 256, 256, 0, stream>>>(a, p);
  spat_att_kernel<<<(BATCH * GRP * NPIX + 255) / 256, 256, 0, stream>>>(p, Wsp, asp);
  scale_pack_kernel<<<eb, 256, 0, stream>>>(a, ach, asp, xm);
  wt_transform_kernel<<<(int)(((size_t)MTOT * KDIM + 255) / 256), 256, 0, stream>>>(W, Wt);
  dim3 grid(NPIX / 64, MTOT / 64, BATCH);
  conv_wmma_kernel<<<grid, 256, 0, stream>>>(xm, Wt, resid, out, add_residual);
}

extern "C" void kernel_launch(void* const* d_in, const int* in_sizes, int n_in,
                              void* d_out, int out_size, void* d_ws, size_t ws_size,
                              hipStream_t stream) {
  const float* x       = (const float*)d_in[0];
  const float* bn1_g   = (const float*)d_in[1];
  const float* bn1_b   = (const float*)d_in[2];
  const float* bn2_g   = (const float*)d_in[3];
  const float* bn2_b   = (const float*)d_in[4];
  const float* c1_W    = (const float*)d_in[5];
  const float* c1_a1   = (const float*)d_in[6];
  const float* c1_a2   = (const float*)d_in[7];
  const float* c1_sp   = (const float*)d_in[8];
  const float* c2_W    = (const float*)d_in[9];
  const float* c2_a1   = (const float*)d_in[10];
  const float* c2_a2   = (const float*)d_in[11];
  const float* c2_sp   = (const float*)d_in[12];

  char* base = (char*)d_ws;
  size_t off = 0;
  auto alloc = [&](size_t bytes) -> void* {
    off = (off + 255) & ~(size_t)255;
    void* pptr = base + off;
    off += bytes;
    return pptr;
  };

  float*  a    = (float*)alloc(NELEM * sizeof(float));
  float*  hbuf = (float*)alloc(NELEM * sizeof(float));
  __bf16* xm   = (__bf16*)alloc(NELEM * sizeof(__bf16));
  __bf16* Wt   = (__bf16*)alloc((size_t)MTOT * KDIM * sizeof(__bf16));
  float*  mean = (float*)alloc(CCH * sizeof(float));
  float*  var  = (float*)alloc(CCH * sizeof(float));
  float*  s    = (float*)alloc(BATCH * CCH * GRP * sizeof(float));
  float*  t    = (float*)alloc(BATCH * MIDC * GRP * sizeof(float));
  float*  ach  = (float*)alloc(BATCH * CCH * GRP * sizeof(float));
  float*  p    = (float*)alloc((size_t)BATCH * 2 * GRP * NPIX * sizeof(float));
  float*  asp  = (float*)alloc((size_t)BATCH * GRP * NPIX * sizeof(float));

  // layer 1: h = f_att_conv_gg(relu(bn(x)))
  run_layer(x, bn1_g, bn1_b, c1_W, c1_a1, c1_a2, c1_sp,
            /*resid=*/x, hbuf, a, xm, Wt, mean, var, s, t, ach, p, asp,
            /*add_residual=*/0, stream);
  // layer 2: out = x + f_att_conv_gg(relu(bn(h)))
  run_layer(hbuf, bn2_g, bn2_b, c2_W, c2_a1, c2_a2, c2_sp,
            /*resid=*/x, (float*)d_out, a, xm, Wt, mean, var, s, t, ach, p, asp,
            /*add_residual=*/1, stream);
}
